// PerformerAttention_8452495638913
// MI455X (gfx1250) — matrix operand order repaired and checked by reading, not compiled, not run on hardware
//
#include <hip/hip_runtime.h>

#define B_  8
#define N_  2048
#define D_  768
#define H_  12
#define DH_ 64
#define M_  256
#define NBH (B_*H_)

typedef __attribute__((ext_vector_type(16))) __bf16 bf16x16;
typedef __attribute__((ext_vector_type(8)))  __bf16 bf16x8;
typedef __attribute__((ext_vector_type(8)))  float  f32x8;

#define WMMA_BF16(a,b,c) __builtin_amdgcn_wmma_f32_16x16x32_bf16(false,(a),false,(b),(short)0,(c),false,false)

// --- CDNA5 async global->LDS copy (16B per lane, tracked by ASYNCcnt) ---
__device__ __forceinline__ void async_b128_to_lds(const void* gptr, void* lptr) {
  uint32_t lds = (uint32_t)(uintptr_t)lptr;     // LDS aperture: addr[31:0] is the LDS byte offset
  uint64_t ga  = (uint64_t)(uintptr_t)gptr;
  asm volatile("global_load_async_to_lds_b128 %0, %1, off" :: "v"(lds), "v"(ga) : "memory");
}
__device__ __forceinline__ void wait_async0() {
  asm volatile("s_wait_asynccnt 0x0" ::: "memory");
}

// ---- fragment loaders: same contiguous k order for A and B (not the ISA table order) ----
// A 16x32 bf16: lane L -> m = L%16, g = L/16; elements: k = [16g..16g+15]
__device__ __forceinline__ bf16x16 frag_a(const __bf16* base, int ldk, int lane) {
  int m = lane & 15, g = lane >> 4;
  const __bf16* row = base + (size_t)m * ldk;
  bf16x8 lo = *(const bf16x8*)(row + 16 * g);
  bf16x8 hi = *(const bf16x8*)(row + 16 * g + 8);
  bf16x16 a;
#pragma unroll
  for (int i = 0; i < 8; ++i) { a[i] = lo[i]; a[8 + i] = hi[i]; }
  return a;
}
// B 32x16 bf16 stored in LDS as [n][k]: lane L -> n = L%16, k = 16*(L/16) + e
__device__ __forceinline__ bf16x16 frag_b(const __bf16* base, int ldk, int lane) {
  int n = lane & 15, g = lane >> 4;
  return *(const bf16x16*)(base + (size_t)n * ldk + 16 * g);
}

// monotone float<->int mapping so integer atomicMax works for signed floats
__device__ __forceinline__ int   ford(float f)  { int i = __float_as_int(f); return (i < 0) ? (i ^ 0x7fffffff) : i; }
__device__ __forceinline__ float funord(int i)  { int j = (i < 0) ? (i ^ 0x7fffffff) : i; return __int_as_float(j); }

// ------------------------------------------------------------------ init
__global__ __launch_bounds__(256) void init_kernel(float* ksum, int* kmaxbits) {
  int bh = blockIdx.x;
  ksum[bh * M_ + threadIdx.x] = 0.f;
  if (threadIdx.x == 0) kmaxbits[bh] = ford(-3.0e38f);
}

// ---------------------------------------------------- QKV projection GEMM
// Y = x @ W^T + b, x:[16384,768] f32, W:[768,768] f32 -> bf16 out in [b,h,n,d]
__global__ __launch_bounds__(256) void qkv_kernel(
    const float* __restrict__ x,
    const float* __restrict__ Wq, const float* __restrict__ bq,
    const float* __restrict__ Wk, const float* __restrict__ bk,
    const float* __restrict__ Wv, const float* __restrict__ bv,
    __bf16* __restrict__ qb, __bf16* __restrict__ kb, __bf16* __restrict__ vb) {
  const float* W; const float* bias; __bf16* out;
  if (blockIdx.z == 0)      { W = Wq; bias = bq; out = qb; }
  else if (blockIdx.z == 1) { W = Wk; bias = bk; out = kb; }
  else                      { W = Wv; bias = bv; out = vb; }

  __shared__ __align__(32) __bf16 As[128][32];
  __shared__ __align__(32) __bf16 Bs[128][32];   // Bs[c][k] = W[colBase+c][k0+k]

  const int tid = threadIdx.x;
  const int wave = tid >> 5, lane = tid & 31;
  const int wm = wave >> 1, wn = wave & 1;       // 4x2 waves -> wave tile 32x64
  const int rowBase = blockIdx.y * 128;
  const int colBase = blockIdx.x * 128;

  f32x8 acc[2][4];
#pragma unroll
  for (int i = 0; i < 2; ++i)
#pragma unroll
    for (int j = 0; j < 4; ++j)
#pragma unroll
      for (int e = 0; e < 8; ++e) acc[i][j][e] = 0.f;

  for (int k0 = 0; k0 < D_; k0 += 32) {
    // prefetch next K-tile into GL2 while this one is consumed (global_prefetch_b8)
    if (k0 + 32 < D_) {
      __builtin_prefetch(x + (size_t)(rowBase + (tid >> 1)) * D_ + k0 + 32, 0, 1);
      __builtin_prefetch(W + (size_t)(colBase + (tid >> 1)) * D_ + k0 + 32, 0, 1);
    }
    __syncthreads();
#pragma unroll
    for (int i = 0; i < 4; ++i) {
      int idx = tid + 256 * i;
      int r = idx >> 3, c4 = (idx & 7) * 4;
      float4 av = *(const float4*)(x + (size_t)(rowBase + r) * D_ + k0 + c4);
      As[r][c4 + 0] = (__bf16)av.x; As[r][c4 + 1] = (__bf16)av.y;
      As[r][c4 + 2] = (__bf16)av.z; As[r][c4 + 3] = (__bf16)av.w;
      float4 wv = *(const float4*)(W + (size_t)(colBase + r) * D_ + k0 + c4);
      Bs[r][c4 + 0] = (__bf16)wv.x; Bs[r][c4 + 1] = (__bf16)wv.y;
      Bs[r][c4 + 2] = (__bf16)wv.z; Bs[r][c4 + 3] = (__bf16)wv.w;
    }
    __syncthreads();
#pragma unroll
    for (int mi = 0; mi < 2; ++mi) {
      bf16x16 a = frag_a(&As[wm * 32 + mi * 16][0], 32, lane);
#pragma unroll
      for (int ni = 0; ni < 4; ++ni) {
        bf16x16 b = frag_b(&Bs[wn * 64 + ni * 16][0], 32, lane);
        acc[mi][ni] = WMMA_BF16(a, b, acc[mi][ni]);
      }
    }
  }
  const int g = lane >> 4, ln = lane & 15;
#pragma unroll
  for (int mi = 0; mi < 2; ++mi)
#pragma unroll
    for (int ni = 0; ni < 4; ++ni)
#pragma unroll
      for (int r = 0; r < 8; ++r) {
        int gr = rowBase + wm * 32 + mi * 16 + 8 * g + r;
        int gc = colBase + wn * 64 + ni * 16 + ln;
        float v = acc[mi][ni][r] + bias[gc];
        int bb = gr >> 11, n = gr & (N_ - 1);      // N_ = 2048
        int h = gc >> 6, dd = gc & 63;
        out[(((size_t)(bb * H_ + h)) * N_ + n) * DH_ + dd] = (__bf16)v;
      }
}

// ----------------------------------------- FAVOR+ feature map (Q fused / K 2-phase)
// data_normalizer is folded into the (f32->bf16 converted) projection matrix, so the
// q/k tile stage is a pure bf16 copy -> async global->LDS.
template <bool IS_Q>
__global__ __launch_bounds__(256) void feat_kernel(
    const __bf16* __restrict__ src,   // [bh][n][64] bf16
    const float*  __restrict__ proj,  // [256][64]   f32
    __bf16* __restrict__ phi,         // [bh][n][256] (Q: phi, K: raw xp)
    float* __restrict__ diagK,        // [bh][n]  (K only)
    int*   __restrict__ kmaxbits) {   // [bh]     (K only)
  __shared__ __align__(32) __bf16 Qs[64][64];
  __shared__ __align__(32) __bf16 Ps[256][32];
  __shared__ __align__(32) __bf16 Xs[64][256];
  __shared__ float diag_s[64];
  __shared__ float red_s[256];

  const int tid = threadIdx.x;
  const int wave = tid >> 5, lane = tid & 31;
  const int wm = wave >> 2, wn = wave & 3;       // 2x4 waves -> wave tile 32x64
  const int bh = blockIdx.y;
  const int n0 = blockIdx.x * 64;
  const float dn  = 0.3535533905932738f;         // 64^-0.25
  const float dn2 = 0.125f;                      // 64^-0.5
  const float ratio = 0.0625f;                   // 256^-0.5

  // stage q/k tile via async LDS DMA (pure copy, ASYNCcnt-tracked)
#pragma unroll
  for (int i = 0; i < 2; ++i) {
    int idx = tid + 256 * i;
    int r = idx >> 3, c8 = (idx & 7) * 8;
    async_b128_to_lds(src + ((size_t)bh * N_ + n0 + r) * DH_ + c8, &Qs[r][c8]);
  }
  wait_async0();
  __syncthreads();
  if (tid < 64) {
    float s = 0.f;
#pragma unroll
    for (int j = 0; j < 64; ++j) { float q = (float)Qs[tid][j]; s += q * q; }
    diag_s[tid] = 0.5f * dn2 * s;
  }

  f32x8 acc[2][4];
#pragma unroll
  for (int i = 0; i < 2; ++i)
#pragma unroll
    for (int j = 0; j < 4; ++j)
#pragma unroll
      for (int e = 0; e < 8; ++e) acc[i][j][e] = 0.f;

  for (int k0 = 0; k0 < 64; k0 += 32) {
    __syncthreads();
#pragma unroll
    for (int i = 0; i < 8; ++i) {                 // Ps[m][k] = dn * proj[m][k0+k]
      int idx = tid + 256 * i;
      int m = idx >> 3, c4 = (idx & 7) * 4;
      float4 pv = *(const float4*)(proj + (size_t)m * DH_ + k0 + c4);
      Ps[m][c4 + 0] = (__bf16)(dn * pv.x); Ps[m][c4 + 1] = (__bf16)(dn * pv.y);
      Ps[m][c4 + 2] = (__bf16)(dn * pv.z); Ps[m][c4 + 3] = (__bf16)(dn * pv.w);
    }
    __syncthreads();
#pragma unroll
    for (int mi = 0; mi < 2; ++mi) {
      bf16x16 a = frag_a(&Qs[wm * 32 + mi * 16][k0], 64, lane);
#pragma unroll
      for (int ni = 0; ni < 4; ++ni) {
        bf16x16 b = frag_b(&Ps[wn * 64 + ni * 16][0], 32, lane);
        acc[mi][ni] = WMMA_BF16(a, b, acc[mi][ni]);
      }
    }
  }
  // spill xp tile to LDS for cross-wave reductions
  const int g = lane >> 4, ln = lane & 15;
  __syncthreads();
#pragma unroll
  for (int mi = 0; mi < 2; ++mi)
#pragma unroll
    for (int ni = 0; ni < 4; ++ni)
#pragma unroll
      for (int r = 0; r < 8; ++r)
        Xs[wm * 32 + mi * 16 + 8 * g + r][wn * 64 + ni * 16 + ln] = (__bf16)acc[mi][ni][r];
  __syncthreads();

  if (IS_Q) {
    if (tid < 64) {                                // per-row max (all 256 cols in-block)
      float m = -3.0e38f;
      for (int j = 0; j < 256; ++j) m = fmaxf(m, (float)Xs[tid][j]);
      red_s[tid] = m;
    }
    __syncthreads();
    for (int e = tid; e < 64 * 256; e += 256) {
      int r = e >> 8, m = e & 255;
      float p = ratio * (__expf((float)Xs[r][m] - diag_s[r] - red_s[r]) + 1e-4f);
      phi[((size_t)bh * N_ + n0 + r) * M_ + m] = (__bf16)p;
    }
  } else {
    float mx = -3.0e38f;
    for (int e = tid; e < 64 * 256; e += 256) {    // store raw xp, track tile max
      int r = e >> 8, c = e & 255;
      float xv = (float)Xs[r][c];
      mx = fmaxf(mx, xv);
      phi[((size_t)bh * N_ + n0 + r) * M_ + c] = (__bf16)xv;
    }
    red_s[tid] = mx;
    __syncthreads();
    for (int s = 128; s > 0; s >>= 1) {
      if (tid < s) red_s[tid] = fmaxf(red_s[tid], red_s[tid + s]);
      __syncthreads();
    }
    if (tid == 0) atomicMax(&kmaxbits[bh], ford(red_s[0]));
    if (tid < 64) diagK[(size_t)bh * N_ + n0 + tid] = diag_s[tid];
  }
}

// ---------------------------------------------- K: exp + ksum accumulation
__global__ __launch_bounds__(256) void kexp_kernel(
    __bf16* __restrict__ phiK, const float* __restrict__ diagK,
    const int* __restrict__ kmaxbits, float* __restrict__ ksum) {
  const int bh = blockIdx.y;
  const int n0 = blockIdx.x * 64;
  const int m = threadIdx.x;
  const float kmax = funord(kmaxbits[bh]);
  float s = 0.f;
  for (int r = 0; r < 64; ++r) {
    size_t off = ((size_t)bh * N_ + n0 + r) * M_ + m;
    float p = 0.0625f * (__expf((float)phiK[off] - diagK[(size_t)bh * N_ + n0 + r] - kmax) + 1e-4f);
    phiK[off] = (__bf16)p;
    s += p;
  }
  atomicAdd(&ksum[bh * M_ + m], s);
}

// ---------------------------------------------- ctx = phiK^T @ v : [bh][256][64]
__global__ __launch_bounds__(256) void ctx_kernel(
    const __bf16* __restrict__ phiK, const __bf16* __restrict__ vb,
    float* __restrict__ ctx) {
  __shared__ __align__(32) __bf16 As[256][40];   // [m][n], padded (80B stride) vs bank hotspot
  __shared__ __align__(32) __bf16 Bs[64][32];    // [dd][n]
  const int tid = threadIdx.x;
  const int wave = tid >> 5, lane = tid & 31;
  const int wm = wave >> 1, wn = wave & 1;       // 4x2 -> wave tile 64x32
  const int bh = blockIdx.x;

  f32x8 acc[4][2];
#pragma unroll
  for (int i = 0; i < 4; ++i)
#pragma unroll
    for (int j = 0; j < 2; ++j)
#pragma unroll
      for (int e = 0; e < 8; ++e) acc[i][j][e] = 0.f;

  for (int n0 = 0; n0 < N_; n0 += 32) {
    __syncthreads();
    for (int i = 0; i < 32; ++i)                  // transposed stage, coalesced on m
      As[tid][i] = phiK[((size_t)bh * N_ + n0 + i) * M_ + tid];
#pragma unroll
    for (int i = 0; i < 8; ++i) {
      int idx = tid + 256 * i;
      int kk = idx >> 6, c = idx & 63;
      Bs[c][kk] = vb[((size_t)bh * N_ + n0 + kk) * DH_ + c];
    }
    __syncthreads();
#pragma unroll
    for (int mi = 0; mi < 4; ++mi) {
      bf16x16 a = frag_a(&As[wm * 64 + mi * 16][0], 40, lane);
#pragma unroll
      for (int ni = 0; ni < 2; ++ni) {
        bf16x16 b = frag_b(&Bs[wn * 32 + ni * 16][0], 32, lane);
        acc[mi][ni] = WMMA_BF16(a, b, acc[mi][ni]);
      }
    }
  }
  const int g = lane >> 4, ln = lane & 15;
#pragma unroll
  for (int mi = 0; mi < 4; ++mi)
#pragma unroll
    for (int ni = 0; ni < 2; ++ni)
#pragma unroll
      for (int r = 0; r < 8; ++r) {
        int gm = wm * 64 + mi * 16 + 8 * g + r;
        int gd = wn * 32 + ni * 16 + ln;
        ctx[((size_t)bh * M_ + gm) * DH_ + gd] = acc[mi][ni][r];
      }
}

// ---------------------------------------------- D_inv = 1 / (phiQ . ksum)
__global__ __launch_bounds__(256) void dinv_kernel(
    const __bf16* __restrict__ phiQ, const float* __restrict__ ksum,
    float* __restrict__ dinv) {
  int row = blockIdx.x * 256 + threadIdx.x;      // row = bh*N + n
  int bh = row >> 11;
  float s = 0.f;
  for (int m = 0; m < M_; ++m)
    s += (float)phiQ[(size_t)row * M_ + m] * ksum[bh * M_ + m];
  dinv[row] = 1.f / s;
}

// ----------------------------- out = (phiQ @ ctx) * D_inv -> attn bf16 [b][n][768]
__global__ __launch_bounds__(256) void attn_kernel(
    const __bf16* __restrict__ phiQ, const float* __restrict__ ctx,
    const float* __restrict__ dinv, __bf16* __restrict__ attn) {
  __shared__ __align__(32) __bf16 Qs[128][32];
  __shared__ __align__(32) __bf16 Bs[64][32];    // [dd][m]
  const int tid = threadIdx.x;
  const int wave = tid >> 5, lane = tid & 31;
  const int wm = wave >> 1, wn = wave & 1;       // 4x2 -> wave tile 32x32
  const int bh = blockIdx.y;
  const int n0 = blockIdx.x * 128;

  f32x8 acc[2][2];
#pragma unroll
  for (int i = 0; i < 2; ++i)
#pragma unroll
    for (int j = 0; j < 2; ++j)
#pragma unroll
      for (int e = 0; e < 8; ++e) acc[i][j][e] = 0.f;

  for (int k0 = 0; k0 < M_; k0 += 32) {
    __syncthreads();
#pragma unroll
    for (int i = 0; i < 2; ++i) {                 // async bf16 tile copy into LDS
      int idx = tid + 256 * i;
      int r = idx >> 2, c8 = (idx & 3) * 8;
      async_b128_to_lds(phiQ + ((size_t)bh * N_ + n0 + r) * M_ + k0 + c8, &Qs[r][c8]);
    }
#pragma unroll
    for (int i = 0; i < 8; ++i) {
      int idx = tid + 256 * i;
      int kk = idx >> 6, c = idx & 63;
      Bs[c][kk] = (__bf16)ctx[((size_t)bh * M_ + k0 + kk) * DH_ + c];
    }
    wait_async0();
    __syncthreads();
#pragma unroll
    for (int mi = 0; mi < 2; ++mi) {
      bf16x16 a = frag_a(&Qs[wm * 32 + mi * 16][0], 32, lane);
#pragma unroll
      for (int ni = 0; ni < 2; ++ni) {
        bf16x16 b = frag_b(&Bs[wn * 32 + ni * 16][0], 32, lane);
        acc[mi][ni] = WMMA_BF16(a, b, acc[mi][ni]);
      }
    }
  }
  const int g = lane >> 4, ln = lane & 15;
  const int bb = bh / H_, h = bh % H_;
#pragma unroll
  for (int mi = 0; mi < 2; ++mi)
#pragma unroll
    for (int ni = 0; ni < 2; ++ni)
#pragma unroll
      for (int r = 0; r < 8; ++r) {
        int n = n0 + wm * 32 + mi * 16 + 8 * g + r;
        int dd = wn * 32 + ni * 16 + ln;
        float v = acc[mi][ni][r] * dinv[(size_t)bh * N_ + n];
        attn[((size_t)bb * N_ + n) * D_ + h * DH_ + dd] = (__bf16)v;
      }
}

// ---------------------------------------------- final projection: attn @ Wo^T + bo
__global__ __launch_bounds__(256) void oproj_kernel(
    const __bf16* __restrict__ attn, const float* __restrict__ Wo,
    const float* __restrict__ bo, float* __restrict__ out) {
  __shared__ __align__(32) __bf16 As[128][32];
  __shared__ __align__(32) __bf16 Bs[128][32];
  const int tid = threadIdx.x;
  const int wave = tid >> 5, lane = tid & 31;
  const int wm = wave >> 1, wn = wave & 1;
  const int rowBase = blockIdx.y * 128;
  const int colBase = blockIdx.x * 128;

  f32x8 acc[2][4];
#pragma unroll
  for (int i = 0; i < 2; ++i)
#pragma unroll
    for (int j = 0; j < 4; ++j)
#pragma unroll
      for (int e = 0; e < 8; ++e) acc[i][j][e] = 0.f;

  for (int k0 = 0; k0 < D_; k0 += 32) {
    if (k0 + 32 < D_)
      __builtin_prefetch(Wo + (size_t)(colBase + (tid >> 1)) * D_ + k0 + 32, 0, 1);
    __syncthreads();
#pragma unroll
    for (int i = 0; i < 2; ++i) {                 // async bf16 tile copy into LDS
      int idx = tid + 256 * i;
      int r = idx >> 2, c8 = (idx & 3) * 8;
      async_b128_to_lds(attn + (size_t)(rowBase + r) * D_ + k0 + c8, &As[r][c8]);
    }
#pragma unroll
    for (int i = 0; i < 4; ++i) {
      int idx = tid + 256 * i;
      int r = idx >> 3, c4 = (idx & 7) * 4;
      float4 w = *(const float4*)(Wo + (size_t)(colBase + r) * D_ + k0 + c4);
      Bs[r][c4 + 0] = (__bf16)w.x; Bs[r][c4 + 1] = (__bf16)w.y;
      Bs[r][c4 + 2] = (__bf16)w.z; Bs[r][c4 + 3] = (__bf16)w.w;
    }
    wait_async0();
    __syncthreads();
#pragma unroll
    for (int mi = 0; mi < 2; ++mi) {
      bf16x16 a = frag_a(&As[wm * 32 + mi * 16][0], 32, lane);
#pragma unroll
      for (int ni = 0; ni < 4; ++ni) {
        bf16x16 b = frag_b(&Bs[wn * 64 + ni * 16][0], 32, lane);
        acc[mi][ni] = WMMA_BF16(a, b, acc[mi][ni]);
      }
    }
  }
  const int g = lane >> 4, ln = lane & 15;
#pragma unroll
  for (int mi = 0; mi < 2; ++mi)
#pragma unroll
    for (int ni = 0; ni < 4; ++ni)
#pragma unroll
      for (int r = 0; r < 8; ++r) {
        int gr = rowBase + wm * 32 + mi * 16 + 8 * g + r;
        int gc = colBase + wn * 64 + ni * 16 + ln;
        out[(size_t)gr * D_ + gc] = acc[mi][ni][r] + bo[gc];
      }
}

// ------------------------------------------------------------------ launch
extern "C" void kernel_launch(void* const* d_in, const int* in_sizes, int n_in,
                              void* d_out, int out_size, void* d_ws, size_t ws_size,
                              hipStream_t stream) {
  (void)in_sizes; (void)n_in; (void)out_size; (void)ws_size;
  const float* x    = (const float*)d_in[0];
  const float* Wq   = (const float*)d_in[1];
  const float* bq   = (const float*)d_in[2];
  const float* Wk   = (const float*)d_in[3];
  const float* bk   = (const float*)d_in[4];
  const float* Wv   = (const float*)d_in[5];
  const float* bv   = (const float*)d_in[6];
  const float* Wo   = (const float*)d_in[7];
  const float* bo   = (const float*)d_in[8];
  const float* proj = (const float*)d_in[9];
  float* out = (float*)d_out;

  char* ws = (char*)d_ws;
  size_t off = 0;
  auto carve = [&](size_t bytes) -> char* {
    char* p = ws + off;
    off = (off + bytes + 255) & ~(size_t)255;
    return p;
  };
  __bf16* qb    = (__bf16*)carve((size_t)NBH * N_ * DH_ * 2);
  __bf16* kb    = (__bf16*)carve((size_t)NBH * N_ * DH_ * 2);
  __bf16* vb    = (__bf16*)carve((size_t)NBH * N_ * DH_ * 2);
  __bf16* phiQ  = (__bf16*)carve((size_t)NBH * N_ * M_ * 2);
  __bf16* phiK  = (__bf16*)carve((size_t)NBH * N_ * M_ * 2);
  float*  diagK = (float*) carve((size_t)NBH * N_ * 4);
  float*  ksum  = (float*) carve((size_t)NBH * M_ * 4);
  int*    kmaxb = (int*)   carve((size_t)NBH * 4);
  float*  ctx   = (float*) carve((size_t)NBH * M_ * DH_ * 4);
  float*  dinv  = (float*) carve((size_t)NBH * N_ * 4);
  __bf16* attn  = (__bf16*)carve((size_t)B_ * N_ * D_ * 2);

  init_kernel<<<NBH, 256, 0, stream>>>(ksum, kmaxb);
  qkv_kernel<<<dim3(D_ / 128, (B_ * N_) / 128, 3), 256, 0, stream>>>(
      x, Wq, bq, Wk, bk, Wv, bv, qb, kb, vb);
  feat_kernel<true ><<<dim3(N_ / 64, NBH), 256, 0, stream>>>(qb, proj, phiQ, (float*)nullptr, (int*)nullptr);
  feat_kernel<false><<<dim3(N_ / 64, NBH), 256, 0, stream>>>(kb, proj, phiK, diagK, kmaxb);
  kexp_kernel<<<dim3(N_ / 64, NBH), 256, 0, stream>>>(phiK, diagK, kmaxb, ksum);
  ctx_kernel<<<NBH, 256, 0, stream>>>(phiK, vb, ctx);
  dinv_kernel<<<(NBH * N_) / 256, 256, 0, stream>>>(phiQ, ksum, dinv);
  attn_kernel<<<dim3(N_ / 128, NBH), 256, 0, stream>>>(phiQ, ctx, dinv, attn);
  oproj_kernel<<<dim3(D_ / 128, (B_ * N_) / 128), 256, 0, stream>>>(attn, Wo, bo, out);
}
